// MultiHeadAttention_1417339208408
// MI455X (gfx1250) — matrix operand order repaired and checked by reading, not compiled, not run on hardware
//
#include <hip/hip_runtime.h>

// ---------------- problem constants ----------------
#define BATCH  4
#define SEQ    2048
#define DMODEL 1024
#define NHEAD  16
#define HDIM   64
#define MROWS  (BATCH * SEQ)   // 8192

typedef __attribute__((ext_vector_type(16))) __bf16 v16bf;
typedef __attribute__((ext_vector_type(8)))  float  v8f;
typedef unsigned int v4u __attribute__((ext_vector_type(4)));
typedef int          v8i __attribute__((ext_vector_type(8)));
typedef int          v4i __attribute__((ext_vector_type(4)));

// native bf16 convert (RNE) -> v_cvt_*bf16_f32 on gfx1250
__device__ __forceinline__ unsigned short f2bf(float x) {
  __bf16 b = (__bf16)x;
  unsigned short u;
  __builtin_memcpy(&u, &b, 2);
  return u;
}

union AFrag { v16bf v; uint4 q[2]; };
union BFrag { v16bf v; uint4 q[2]; };
union CTile { v8f  v; float f[8]; };

// ---- Tensor Data Mover: 2D tile (bf16, data_size=2B) global -> LDS ----
// D# layout per CDNA5 ISA 8.3/8.4: group0 = {ctl, lds_addr, global_addr, type},
// group1 = {mask/data_size, tensor_dim0/1, tile_dim0/1/2, dim0_stride, dim1_stride}.
// This toolchain exposes the 6-arg builtin: (v4u, v8i, v4i, v4i, v8i, cpol).
__device__ __forceinline__ void tdm_load_2d_bf16(
    const void* gptr, unsigned lds_off,
    unsigned tensor_d0, unsigned tensor_d1,
    unsigned tile_d0, unsigned tile_d1, unsigned stride0)
{
  unsigned long long ga = (unsigned long long)gptr;
  v4u g0;
  g0[0] = 1u;                                              // count=1, user D#
  g0[1] = lds_off;                                         // LDS byte address
  g0[2] = (unsigned)ga;                                    // global_addr[31:0]
  g0[3] = (unsigned)((ga >> 32) & 0x01FFFFFFull) | (2u << 30);  // [56:32] | type=2
  v8i g1;
  g1[0] = 0x00010000;                                      // workgroup_mask=0, data_size=1 (2B)
  g1[1] = (int)(tensor_d0 << 16);                          // tensor_dim0 lo16 @ [31:16]
  g1[2] = (int)((tensor_d0 >> 16) | (tensor_d1 << 16));    // t_dim0 hi | t_dim1 lo
  g1[3] = (int)((tensor_d1 >> 16) | (tile_d0 << 16));      // t_dim1 hi | tile_dim0
  g1[4] = (int)(tile_d1 & 0xFFFFu);                        // tile_dim1 | tile_dim2=0
  g1[5] = (int)stride0;                                    // tensor_dim0_stride lo32
  g1[6] = 0;                                               // stride0 hi | dim1_stride lo
  g1[7] = 0;                                               // dim1_stride hi
  v4i z4 = {};
  v8i z8 = {};
  __builtin_amdgcn_tensor_load_to_lds(g0, g1, z4, z4, z8, 0);
}

// C = A[M,K] * W[N,K]^T  (M=8192, N=K=1024)
// OUT_MODE 0: bf16 [B,H,L,HD]   (Q, K)
// OUT_MODE 1: bf16 [B,H,HD,L]   (V, transposed per head)
// OUT_MODE 2: f32  plain [M,N]  (final output)
template <int OUT_MODE, bool A_BF16>
__global__ __launch_bounds__(256) void gemm_bf16_wmma(
    const void* __restrict__ Aptr, const float* __restrict__ Wptr,
    void* __restrict__ Cptr)
{
  __shared__ __align__(16) unsigned short As[128][32];
  __shared__ __align__(16) unsigned short Bs[64][32];

  const int tid  = threadIdx.x;
  const int lane = tid & 31;
  const int wv   = tid >> 5;      // 0..7 waves
  const int wm   = wv >> 1;       // 0..3  -> 32-row slab
  const int wn   = wv & 1;        // 0..1  -> 32-col slab
  const int m0   = blockIdx.x * 128;
  const int n0   = blockIdx.y * 64;

  CTile acc[2][2];
  #pragma unroll
  for (int i = 0; i < 2; ++i)
    #pragma unroll
    for (int j = 0; j < 2; ++j)
      #pragma unroll
      for (int r = 0; r < 8; ++r) acc[i][j].f[r] = 0.0f;

  const int abase = (lane < 16) ? 0 : 16;  // A-operand half-wave K offset
  const int arow  = lane & 15;
  const int bkoff = (lane < 16) ? 0 : 16;  // B-operand half-wave K offset
  const int bcol  = lane & 15;

  for (int kt = 0; kt < DMODEL / 32; ++kt) {
    // ---- stage A tile (128 x 32) into LDS as bf16 ----
    if (A_BF16) {
      const unsigned short* A = (const unsigned short*)Aptr;
      #pragma unroll
      for (int i = 0; i < 2; ++i) {
        int idx = tid + i * 256;   // uint4 index (8 bf16), 4 per row
        int row = idx >> 2;
        int c8  = (idx & 3) * 8;
        *(uint4*)&As[row][c8] =
            *(const uint4*)(A + (size_t)(m0 + row) * DMODEL + kt * 32 + c8);
      }
    } else {
      const float* A = (const float*)Aptr;
      #pragma unroll
      for (int i = 0; i < 4; ++i) {
        int idx = tid + i * 256;   // float4 index, 8 per row
        int row = idx >> 3;
        int c4  = (idx & 7) * 4;
        float4 vv = *(const float4*)(A + (size_t)(m0 + row) * DMODEL + kt * 32 + c4);
        uint2 p;
        p.x = (unsigned)f2bf(vv.x) | ((unsigned)f2bf(vv.y) << 16);
        p.y = (unsigned)f2bf(vv.z) | ((unsigned)f2bf(vv.w) << 16);
        *(uint2*)&As[row][c4] = p;
      }
    }
    // ---- stage B tile: W rows n0..n0+63, cols kt*32..+31 ----
    #pragma unroll
    for (int i = 0; i < 2; ++i) {
      int idx = tid + i * 256;     // float4 index, 8 per row
      int row = idx >> 3;
      int c4  = (idx & 7) * 4;
      float4 vv = *(const float4*)(Wptr + (size_t)(n0 + row) * DMODEL + kt * 32 + c4);
      uint2 p;
      p.x = (unsigned)f2bf(vv.x) | ((unsigned)f2bf(vv.y) << 16);
      p.y = (unsigned)f2bf(vv.z) | ((unsigned)f2bf(vv.w) << 16);
      *(uint2*)&Bs[row][c4] = p;
    }
    if (kt + 1 < DMODEL / 32) {    // warm GL2 for next K-slice
      __builtin_prefetch((const char*)Wptr + ((size_t)n0 * DMODEL + (kt + 1) * 32) * 4, 0, 1);
    }
    __syncthreads();

    // ---- fragments from LDS in contiguous K order (k = 16*half + s), same for A and B ----
    AFrag a[2];
    BFrag bq[2];
    #pragma unroll
    for (int i = 0; i < 2; ++i) {
      const unsigned short* rp = &As[wm * 32 + i * 16 + arow][0];
      a[i].q[0] = *(const uint4*)(rp + abase);        // K = abase .. abase+7
      a[i].q[1] = *(const uint4*)(rp + abase + 8);    // K = abase+8 .. abase+15
    }
    #pragma unroll
    for (int j = 0; j < 2; ++j) {
      const unsigned short* rp = &Bs[wn * 32 + j * 16 + bcol][0];
      bq[j].q[0] = *(const uint4*)(rp + bkoff);       // K = bkoff .. bkoff+15
      bq[j].q[1] = *(const uint4*)(rp + bkoff + 8);
    }
    #pragma unroll
    for (int i = 0; i < 2; ++i)
      #pragma unroll
      for (int j = 0; j < 2; ++j)
        acc[i][j].v = __builtin_amdgcn_wmma_f32_16x16x32_bf16(
            false, a[i].v, false, bq[j].v, (short)0, acc[i][j].v, false, false);
    __syncthreads();
  }

  // ---- epilogue: C/D layout -> memory ----
  const int coff = (lane >> 4) * 8;  // rows 0..7 (lanes 0-15), 8..15 (lanes 16-31)
  const int ccol = lane & 15;
  #pragma unroll
  for (int i = 0; i < 2; ++i) {
    #pragma unroll
    for (int j = 0; j < 2; ++j) {
      #pragma unroll
      for (int r = 0; r < 8; ++r) {
        int m = m0 + wm * 32 + i * 16 + coff + r;
        int n = n0 + wn * 32 + j * 16 + ccol;
        float val = acc[i][j].f[r];
        if (OUT_MODE == 2) {
          ((float*)Cptr)[(size_t)m * DMODEL + n] = val;
        } else {
          int bb = m >> 11, l = m & (SEQ - 1);
          int hh = n >> 6,  d = n & (HDIM - 1);
          size_t idx = (OUT_MODE == 0)
              ? ((((size_t)bb * NHEAD + hh) * SEQ + l) * HDIM + d)
              : ((((size_t)bb * NHEAD + hh) * HDIM + d) * SEQ + l);
          ((unsigned short*)Cptr)[idx] = f2bf(val);
        }
      }
    }
  }
}

// Fused flash-style attention: one block = 128 query rows of one (b,h).
// 8 waves x 16 query rows; K/V blocks of 64 double-buffered through LDS by the
// Tensor Data Mover (DMA overlapped with WMMA compute), online softmax.
__global__ __launch_bounds__(256) void attn_fused_wmma(
    const unsigned short* __restrict__ Qp,   // [B,H,L,HD] bf16
    const unsigned short* __restrict__ Kp,   // [B,H,L,HD] bf16
    const unsigned short* __restrict__ Vt,   // [B,H,HD,L] bf16 (transposed)
    unsigned short* __restrict__ Ctx)        // [B,L,D]    bf16
{
  __shared__ __align__(16) unsigned short Ks[2][64][64];   // [buf][j][d]
  __shared__ __align__(16) unsigned short Vs[2][64][64];   // [buf][d][j]
  __shared__ __align__(16) unsigned short Ps[8][16][64];   // per-wave P tile

  const int tid = threadIdx.x, lane = tid & 31, wv = tid >> 5;
  const int bh = blockIdx.y;
  const int b  = bh >> 4;            // / NHEAD
  const int h  = bh & (NHEAD - 1);
  const int q0 = blockIdx.x * 128;

  const size_t hoff = (size_t)bh * SEQ * HDIM;
  const unsigned short* Qh = Qp + hoff;
  const unsigned short* Kh = Kp + hoff;
  const unsigned short* Vh = Vt + hoff;

  const unsigned ks_lds0 = (unsigned)(unsigned long long)(void*)&Ks[0][0][0];
  const unsigned ks_lds1 = (unsigned)(unsigned long long)(void*)&Ks[1][0][0];
  const unsigned vs_lds0 = (unsigned)(unsigned long long)(void*)&Vs[0][0][0];
  const unsigned vs_lds1 = (unsigned)(unsigned long long)(void*)&Vs[1][0][0];

  const int abase = (lane < 16) ? 0 : 16;
  const int arow  = lane & 15;
  const int bkoff = (lane < 16) ? 0 : 16;
  const int bcol  = lane & 15;
  const int coff  = (lane >> 4) * 8;

  // Q fragments live in registers for the whole kernel
  AFrag qa[2];
  {
    const unsigned short* qr = Qh + (size_t)(q0 + wv * 16 + arow) * HDIM;
    #pragma unroll
    for (int ks = 0; ks < 2; ++ks) {
      qa[ks].q[0] = *(const uint4*)(qr + ks * 32 + abase);
      qa[ks].q[1] = *(const uint4*)(qr + ks * 32 + abase + 8);
    }
  }

  CTile o[4];
  #pragma unroll
  for (int t = 0; t < 4; ++t)
    #pragma unroll
    for (int r = 0; r < 8; ++r) o[t].f[r] = 0.0f;
  float mrow[8], lrow[8];
  #pragma unroll
  for (int r = 0; r < 8; ++r) { mrow[r] = -3.0e38f; lrow[r] = 0.0f; }

  const float scale = 0.125f;  // 1/sqrt(HD)

  // prologue: kick off DMA of the first K/V block into buffer 0
  if (wv == 0) {
    tdm_load_2d_bf16(Kh, ks_lds0, HDIM, SEQ, HDIM, 64, HDIM);
    tdm_load_2d_bf16(Vh, vs_lds0, SEQ, HDIM, 64, HDIM, SEQ);
  }

  for (int jb = 0; jb < SEQ; jb += 64) {
    const int cur = (jb >> 6) & 1;
    // ---- issue next block's DMA into the other buffer, then wait for cur ----
    if (wv == 0) {
      if (jb + 64 < SEQ) {
        const int nj = jb + 64;
        tdm_load_2d_bf16(Kh + (size_t)nj * HDIM, cur ? ks_lds0 : ks_lds1,
                         HDIM, SEQ, HDIM, 64, HDIM);
        tdm_load_2d_bf16(Vh + nj, cur ? vs_lds0 : vs_lds1,
                         SEQ, HDIM, 64, HDIM, SEQ);
        // TENSORcnt completes in-order: <=2 outstanding => current block done
        __builtin_amdgcn_s_wait_tensorcnt(2);
      } else {
        __builtin_amdgcn_s_wait_tensorcnt(0);
      }
    }
    __syncthreads();

    // ---- S = Q * K^T  (16x64 per wave, f32 accum) ----
    CTile s[4];
    #pragma unroll
    for (int t = 0; t < 4; ++t)
      #pragma unroll
      for (int r = 0; r < 8; ++r) s[t].f[r] = 0.0f;
    #pragma unroll
    for (int ks = 0; ks < 2; ++ks) {
      #pragma unroll
      for (int t = 0; t < 4; ++t) {
        BFrag kb;
        const unsigned short* rp = &Ks[cur][t * 16 + bcol][ks * 32 + bkoff];
        kb.q[0] = *(const uint4*)rp;
        kb.q[1] = *(const uint4*)(rp + 8);
        s[t].v = __builtin_amdgcn_wmma_f32_16x16x32_bf16(
            false, qa[ks].v, false, kb.v, (short)0, s[t].v, false, false);
      }
    }

    // ---- online softmax; rows live across 16-lane half-waves ----
    #pragma unroll
    for (int t = 0; t < 4; ++t)
      #pragma unroll
      for (int r = 0; r < 8; ++r) s[t].f[r] *= scale;

    #pragma unroll
    for (int r = 0; r < 8; ++r) {
      float mx = fmaxf(fmaxf(s[0].f[r], s[1].f[r]), fmaxf(s[2].f[r], s[3].f[r]));
      #pragma unroll
      for (int sh = 1; sh < 16; sh <<= 1) mx = fmaxf(mx, __shfl_xor(mx, sh, 32));
      float mn = fmaxf(mrow[r], mx);
      float su = 0.0f;
      #pragma unroll
      for (int t = 0; t < 4; ++t) {
        float p = __expf(s[t].f[r] - mn);
        s[t].f[r] = p;
        su += p;
      }
      #pragma unroll
      for (int sh = 1; sh < 16; sh <<= 1) su += __shfl_xor(su, sh, 32);
      float alpha = __expf(mrow[r] - mn);
      lrow[r] = lrow[r] * alpha + su;
      mrow[r] = mn;
      #pragma unroll
      for (int t = 0; t < 4; ++t) o[t].f[r] *= alpha;
    }

    // ---- P -> per-wave LDS tile so it can re-enter WMMA as the A operand ----
    #pragma unroll
    for (int r = 0; r < 8; ++r)
      #pragma unroll
      for (int t = 0; t < 4; ++t)
        Ps[wv][coff + r][(lane & 15) + t * 16] = f2bf(s[t].f[r]);
    // same-wave DS ordering: drain ds stores before reloading P
    asm volatile("s_wait_dscnt 0x0" ::: "memory");

    // ---- O += P * V ----
    #pragma unroll
    for (int ks = 0; ks < 2; ++ks) {
      AFrag pa;
      const unsigned short* pp = &Ps[wv][arow][ks * 32];
      pa.q[0] = *(const uint4*)(pp + abase);
      pa.q[1] = *(const uint4*)(pp + abase + 8);
      #pragma unroll
      for (int t = 0; t < 4; ++t) {
        BFrag vb;
        const unsigned short* rp = &Vs[cur][t * 16 + bcol][ks * 32 + bkoff];
        vb.q[0] = *(const uint4*)rp;
        vb.q[1] = *(const uint4*)(rp + 8);
        o[t].v = __builtin_amdgcn_wmma_f32_16x16x32_bf16(
            false, pa.v, false, vb.v, (short)0, o[t].v, false, false);
      }
    }
    __syncthreads();   // all waves done with 'cur' before its buffer is re-filled
  }

  // ---- normalize and write context [B, L, D] (bf16) ----
  #pragma unroll
  for (int r = 0; r < 8; ++r) {
    float inv = 1.0f / lrow[r];
    int qrow = q0 + wv * 16 + coff + r;
    #pragma unroll
    for (int t = 0; t < 4; ++t) {
      int dg = h * HDIM + (lane & 15) + t * 16;
      Ctx[((size_t)b * SEQ + qrow) * DMODEL + dg] = f2bf(o[t].f[r] * inv);
    }
  }
}

extern "C" void kernel_launch(void* const* d_in, const int* in_sizes, int n_in,
                              void* d_out, int out_size, void* d_ws, size_t ws_size,
                              hipStream_t stream) {
  (void)in_sizes; (void)n_in; (void)out_size; (void)ws_size;
  const float* q  = (const float*)d_in[0];
  const float* k  = (const float*)d_in[1];
  const float* v  = (const float*)d_in[2];
  const float* Wq = (const float*)d_in[3];
  const float* Wk = (const float*)d_in[4];
  const float* Wv = (const float*)d_in[5];
  const float* Wo = (const float*)d_in[6];

  const size_t HEAD_ELEMS = (size_t)BATCH * NHEAD * SEQ * HDIM;  // 8Mi
  unsigned short* Qp  = (unsigned short*)d_ws;
  unsigned short* Kp  = Qp + HEAD_ELEMS;
  unsigned short* Vt  = Kp + HEAD_ELEMS;
  unsigned short* Ctx = Vt + HEAD_ELEMS;

  dim3 blk(256);
  dim3 gg(MROWS / 128, DMODEL / 64);   // 64 x 16 blocks

  gemm_bf16_wmma<0, false><<<gg, blk, 0, stream>>>(q, Wq, Qp);
  gemm_bf16_wmma<0, false><<<gg, blk, 0, stream>>>(k, Wk, Kp);
  gemm_bf16_wmma<1, false><<<gg, blk, 0, stream>>>(v, Wv, Vt);
  attn_fused_wmma<<<dim3(SEQ / 128, BATCH * NHEAD), blk, 0, stream>>>(Qp, Kp, Vt, Ctx);
  gemm_bf16_wmma<2, true><<<gg, blk, 0, stream>>>(Ctx, Wo, (float*)d_out);
}
